// CrossTransformer_72748156060364
// MI455X (gfx1250) — matrix operand order repaired and checked by reading, not compiled, not run on hardware
//
#include <hip/hip_runtime.h>

// Problem constants (from reference)
#define BATCH   32
#define NSUP    25
#define CDIM    512
#define PVAL    196     // H*W
#define PPAD    208     // 13 * 16
#define DDIM    128
#define MPERN   208     // padded positions per support image
#define MVALID  5200    // NSUP * MPERN
#define MPAD    5216    // 163 * 32
#define MSTEPS  163     // MPAD / 32
#define MTILES  326     // MPAD / 16

typedef __attribute__((ext_vector_type(16))) __bf16 v16bf;
typedef __attribute__((ext_vector_type(8)))  float  v8f;
typedef unsigned short u16;
typedef unsigned int   u32;

__device__ __forceinline__ u16 f2bf(float f) {
  u32 u = __float_as_uint(f);
  u32 r = (u + 0x7FFFu + ((u >> 16) & 1u)) >> 16;   // round-to-nearest-even
  return (u16)r;
}

union BF16x16 { uint4 q[2]; v16bf v; };

// A-fragment (16x32 bf16): lane holds row M=lane%16; 16 contiguous K [c0+(c0&8) .. +15]
__device__ __forceinline__ v16bf ld_a_frag(const u16* row, int c0) {
  BF16x16 t;
  t.q[0] = *reinterpret_cast<const uint4*>(row + c0 + (c0 & 8));
  t.q[1] = *reinterpret_cast<const uint4*>(row + c0 + (c0 & 8) + 8);
  return t.v;
}
// B-fragment (32x16 bf16): lane holds col N=lane%16; 16 contiguous K values
__device__ __forceinline__ v16bf ld_b_frag(const u16* p) {
  BF16x16 t;
  t.q[0] = *reinterpret_cast<const uint4*>(p);
  t.q[1] = *reinterpret_cast<const uint4*>(p + 8);
  return t.v;
}
__device__ __forceinline__ uint4 pack8(v8f a) {
  uint4 r;
  r.x = (u32)f2bf(a[0]) | ((u32)f2bf(a[1]) << 16);
  r.y = (u32)f2bf(a[2]) | ((u32)f2bf(a[3]) << 16);
  r.z = (u32)f2bf(a[4]) | ((u32)f2bf(a[5]) << 16);
  r.w = (u32)f2bf(a[6]) | ((u32)f2bf(a[7]) << 16);
  return r;
}

#define WMMA_BF16(A, Bm, Cm) \
  __builtin_amdgcn_wmma_f32_16x16x32_bf16(false, (A), false, (Bm), (short)0, (Cm), false, false)

// ---------------- conversion kernels (f32 -> padded bf16) ----------------

// Supports: LDS-staged transpose. Reads float4-coalesced along ij, writes
// dword-coalesced along c. Emits padding rows (ij in [196,208)) as zeros.
#define TSTRIDE 36   // 32 c + 4 pad (u16), keeps LDS banks staggered
__global__ __launch_bounds__(256) void conv_s_kernel(const float* __restrict__ sup,
                                                     u16* __restrict__ Sb) {
  __shared__ __align__(16) u16 tile[PVAL * TSTRIDE];
  const int b  = blockIdx.z;
  const int n  = blockIdx.y;
  const int c0 = blockIdx.x * 32;

  const float* src = sup + (((size_t)b * NSUP + n) * CDIM + c0) * PVAL;
  for (int i = threadIdx.x; i < 32 * 49; i += 256) {     // 32 c-rows x 49 float4
    const int c = i / 49, ij4 = i - c * 49;
    float4 v = *reinterpret_cast<const float4*>(src + (size_t)c * PVAL + ij4 * 4);
    u16* t0 = &tile[(ij4 * 4) * TSTRIDE + c];
    t0[0]           = f2bf(v.x);
    t0[TSTRIDE]     = f2bf(v.y);
    t0[2 * TSTRIDE] = f2bf(v.z);
    t0[3 * TSTRIDE] = f2bf(v.w);
  }
  __syncthreads();
  u16* dst = Sb + ((size_t)b * MPAD + (size_t)n * MPERN) * CDIM + c0;
  for (int i = threadIdx.x; i < PPAD * 16; i += 256) {   // 208 ij-rows x 16 dwords
    const int ij = i >> 4, k = i & 15;
    u32 val = (ij < PVAL) ? *reinterpret_cast<const u32*>(&tile[ij * TSTRIDE + 2 * k]) : 0u;
    *reinterpret_cast<u32*>(dst + (size_t)ij * CDIM + 2 * k) = val;
  }
}

// Zero the global tail m in [MVALID, MPAD) of Sb (tiny: 16 rows per batch)
__global__ void zero_tail_kernel(u16* __restrict__ Sb) {
  const int per_b = 16 * CDIM / 8;                       // uint4 stores per batch
  int t = blockIdx.x * 256 + threadIdx.x;                // BATCH*per_b = 32768 threads
  int b = t / per_b, r = t - b * per_b;
  if (b >= BATCH) return;
  *reinterpret_cast<uint4*>(Sb + ((size_t)b * MPAD + MVALID) * CDIM + (size_t)r * 8) =
      make_uint4(0u, 0u, 0u, 0u);
}

__global__ void conv_q_kernel(const float* __restrict__ q, u16* __restrict__ Qb) {
  size_t t = (size_t)blockIdx.x * 256 + threadIdx.x;
  const size_t TOT = (size_t)BATCH * CDIM * 49;
  if (t >= TOT) return;
  int ij4 = (int)(t % 49);  size_t r = t / 49;
  int c = (int)(r % CDIM);  int b = (int)(r / CDIM);
  float4 v = *reinterpret_cast<const float4*>(
      q + ((size_t)b * CDIM + c) * PVAL + (size_t)ij4 * 4);
  size_t base = ((size_t)b * PPAD + (size_t)ij4 * 4) * CDIM + c;
  Qb[base]            = f2bf(v.x);
  Qb[base + CDIM]     = f2bf(v.y);
  Qb[base + 2 * CDIM] = f2bf(v.z);
  Qb[base + 3 * CDIM] = f2bf(v.w);
}

__global__ void conv_w_kernel(const float* __restrict__ Wqk, const float* __restrict__ Wv,
                              u16* __restrict__ WkB, u16* __restrict__ WvB) {
  int t = blockIdx.x * 256 + threadIdx.x;          // 32768 threads
  const float* src = (t < 16384) ? Wqk : Wv;
  u16* dst = (t < 16384) ? WkB : WvB;
  int r = t & 16383;
  float4 v = *reinterpret_cast<const float4*>(src + (size_t)r * 4);
  u32 lo = (u32)f2bf(v.x) | ((u32)f2bf(v.y) << 16);
  u32 hi = (u32)f2bf(v.z) | ((u32)f2bf(v.w) << 16);
  *reinterpret_cast<uint2*>(dst + (size_t)r * 4) = make_uint2(lo, hi);
}

// ---------------- support projections: Sk[b][m][d], Sv[b][d][m] ----------------

__global__ __launch_bounds__(256) void proj_supports_kernel(
    const u16* __restrict__ Sb, const u16* __restrict__ Wk, const u16* __restrict__ Wv,
    u16* __restrict__ Sk, u16* __restrict__ Sv) {
  const int lane = threadIdx.x & 31;
  const int wave = threadIdx.x >> 5;
  const int l16 = lane & 15, half = lane >> 4;
  const int mt = blockIdx.x * 8 + wave;
  if (mt >= MTILES) return;
  const int dt = blockIdx.y;
  const int b  = blockIdx.z;
  const int m16 = mt * 16, d16 = dt * 16;

  const u16* sbrow = Sb + ((size_t)b * MPAD + m16 + l16) * CDIM;  // m row (A) / m col (B)
  const u16* wkrow = Wk + (size_t)(d16 + l16) * CDIM;
  const u16* wvrow = Wv + (size_t)(d16 + l16) * CDIM;

  v8f accK = {}, accV = {};
#pragma unroll 4
  for (int k = 0; k < 16; ++k) {
    const int c = 32 * k;
    // Sk = Wqk[D,C] x S[C,M]  (A rows = d, B cols = m)
    v16bf aW = ld_a_frag(wkrow, c + (half ? 8 : 0));
    v16bf bS = ld_b_frag(sbrow + c + (half ? 16 : 0));
    accK = WMMA_BF16(aW, bS, accK);
    // Sv^T = S^T[M,C] x Wv^T[C,D]  (A rows = m, B cols = d)
    v16bf aS = ld_a_frag(sbrow, c + (half ? 8 : 0));
    v16bf bW = ld_b_frag(wvrow + c + (half ? 16 : 0));
    accV = WMMA_BF16(aS, bW, accV);
  }
  // Sk D-frag: lane col m fixed, rows d contiguous -> [m][d] layout store
  *reinterpret_cast<uint4*>(Sk + ((size_t)b * MPAD + m16 + l16) * DDIM + d16 + 8 * half) = pack8(accK);
  // Sv D-frag: lane col d fixed, rows m contiguous -> [d][m] layout store
  *reinterpret_cast<uint4*>(Sv + ((size_t)b * DDIM + d16 + l16) * MPAD + m16 + 8 * half) = pack8(accV);
}

// ---------------- query projections: Qq bf16 [b][p][d], Qv f32 [b][p][d] ----------------

__global__ __launch_bounds__(64) void proj_query_kernel(
    const u16* __restrict__ Qb, const u16* __restrict__ Wk, const u16* __restrict__ Wv,
    u16* __restrict__ Qq, float* __restrict__ Qv) {
  const int lane = threadIdx.x & 31;
  const int wave = threadIdx.x >> 5;           // 0 -> Qq (Wqk), 1 -> Qv (Wv)
  const int l16 = lane & 15, half = lane >> 4;
  const int pt = blockIdx.x, dt = blockIdx.y, b = blockIdx.z;
  const int p16 = pt * 16, d16 = dt * 16;

  const u16* wrow = (wave ? Wv : Wk) + (size_t)(d16 + l16) * CDIM;
  const u16* qrow = Qb + ((size_t)b * PPAD + p16 + l16) * CDIM;

  v8f acc = {};
#pragma unroll 4
  for (int k = 0; k < 16; ++k) {
    const int c = 32 * k;
    v16bf aW = ld_a_frag(wrow, c + (half ? 8 : 0));
    v16bf bQ = ld_b_frag(qrow + c + (half ? 16 : 0));
    acc = WMMA_BF16(aW, bQ, acc);
  }
  if (wave == 0) {
    *reinterpret_cast<uint4*>(Qq + ((size_t)b * PPAD + p16 + l16) * DDIM + d16 + 8 * half) = pack8(acc);
  } else {
    float* dst = Qv + ((size_t)b * PPAD + p16 + l16) * DDIM + d16 + 8 * half;
    *reinterpret_cast<float4*>(dst)     = make_float4(acc[0], acc[1], acc[2], acc[3]);
    *reinterpret_cast<float4*>(dst + 4) = make_float4(acc[4], acc[5], acc[6], acc[7]);
  }
}

// ---------------- fused attention + distance (flash-style online softmax) ----------------

__global__ __launch_bounds__(256) void attn_out_kernel(
    const u16* __restrict__ Qq, const float* __restrict__ Qv,
    const u16* __restrict__ Sk, const u16* __restrict__ Sv,
    float* __restrict__ out_scalar) {
  __shared__ __align__(16) float O_sh[16 * DDIM];
  __shared__ __align__(16) u16 probs_sh[8][16 * 32];
  __shared__ float wmaxs[8][16], wsums[8][16];
  __shared__ float gmax[16], gsum[16];
  __shared__ float wred[8];

  const int b    = blockIdx.y;
  const int pt   = blockIdx.x;
  const int lane = threadIdx.x & 31;
  const int wave = threadIdx.x >> 5;
  const int l16  = lane & 15;
  const int half = lane >> 4;

  for (int i = threadIdx.x; i < 16 * DDIM; i += 256) O_sh[i] = 0.f;

  // Qq A-fragments for this p-tile (K = D = 128 -> 4 fragments), held in registers
  const u16* qrow = Qq + ((size_t)b * PPAD + pt * 16 + l16) * DDIM;
  v16bf aq[4];
#pragma unroll
  for (int k = 0; k < 4; ++k) aq[k] = ld_a_frag(qrow, 32 * k + (half ? 8 : 0));

  const u16* skbase = Sk + (size_t)b * MPAD * DDIM;
  const u16* svbase = Sv + (size_t)b * DDIM * MPAD;

  float rm[8], rs[8];
  v8f o[8] = {};
#pragma unroll
  for (int v = 0; v < 8; ++v) { rm[v] = -3.0e38f; rs[v] = 0.f; }

  for (int it = wave; it < MSTEPS; it += 8) {
    const int m0 = it * 32;
    if (it + 8 < MSTEPS) {
      __builtin_prefetch(skbase + (size_t)(m0 + 256 + l16) * DDIM, 0, 1);
      __builtin_prefetch(svbase + (size_t)l16 * MPAD + m0 + 256, 0, 1);
    }
    // sim tiles for columns [m0, m0+16) and [m0+16, m0+32)
    v8f s0 = {}, s1 = {};
    const u16* kr0 = skbase + (size_t)(m0 + l16) * DDIM + (half ? 16 : 0);
    const u16* kr1 = kr0 + (size_t)16 * DDIM;
#pragma unroll
    for (int k = 0; k < 4; ++k) {
      v16bf b0 = ld_b_frag(kr0 + 32 * k);
      s0 = WMMA_BF16(aq[k], b0, s0);
      v16bf b1 = ld_b_frag(kr1 + 32 * k);
      s1 = WMMA_BF16(aq[k], b1, s1);
    }
    // padding mask (per-lane uniform: lane owns one column per tile)
    const int mA = m0 + l16, mB = mA + 16;
    const bool vA = (mA < MVALID) && ((mA % MPERN) < PVAL);
    const bool vB = (mB < MVALID) && ((mB % MPERN) < PVAL);
    const float NEGI = -3.0e38f;
    float nm[8];
#pragma unroll
    for (int v = 0; v < 8; ++v) {
      float a = vA ? s0[v] : NEGI;
      float c = vB ? s1[v] : NEGI;
      s0[v] = a; s1[v] = c;
      nm[v] = fmaxf(a, c);
    }
    // row max across the 16 lanes of each half (rows = v + 8*half)
#pragma unroll
    for (int d = 1; d < 16; d <<= 1)
#pragma unroll
      for (int v = 0; v < 8; ++v) nm[v] = fmaxf(nm[v], __shfl_xor(nm[v], d, 16));

    float scale[8], pA[8], pB[8], rsum[8];
#pragma unroll
    for (int v = 0; v < 8; ++v) {
      nm[v] = fmaxf(nm[v], rm[v]);
      scale[v] = __expf(rm[v] - nm[v]);
      rm[v] = nm[v];
      pA[v] = vA ? __expf(s0[v] - nm[v]) : 0.f;
      pB[v] = vB ? __expf(s1[v] - nm[v]) : 0.f;
      rsum[v] = pA[v] + pB[v];
    }
#pragma unroll
    for (int d = 1; d < 16; d <<= 1)
#pragma unroll
      for (int v = 0; v < 8; ++v) rsum[v] += __shfl_xor(rsum[v], d, 16);
#pragma unroll
    for (int v = 0; v < 8; ++v) rs[v] = rs[v] * scale[v] + rsum[v];
    // rescale running output (row index = v + 8*half is lane-local: per-element scale)
#pragma unroll
    for (int t = 0; t < 8; ++t)
#pragma unroll
      for (int v = 0; v < 8; ++v) o[t][v] *= scale[v];

    // transpose probs (D-frag: lane=col) -> A-frag (lane=row) through LDS
    u16* pw = probs_sh[wave];
#pragma unroll
    for (int v = 0; v < 8; ++v) {
      const int r = v + 8 * half;
      pw[r * 32 + l16]      = f2bf(pA[v]);
      pw[r * 32 + 16 + l16] = f2bf(pB[v]);
    }
    v16bf ap = ld_a_frag(pw + l16 * 32, half ? 8 : 0);
    // O[p, d] += P~[p, m(32)] x Sv[m(32), d] over 8 d-tiles
#pragma unroll
    for (int t = 0; t < 8; ++t) {
      const u16* svp = svbase + (size_t)(t * 16 + l16) * MPAD + m0 + (half ? 16 : 0);
      v16bf bs = ld_b_frag(svp);
      o[t] = WMMA_BF16(ap, bs, o[t]);
    }
  }

  // publish per-wave row stats
  if (l16 == 0) {
#pragma unroll
    for (int v = 0; v < 8; ++v) {
      const int r = v + 8 * half;
      wmaxs[wave][r] = rm[v];
      wsums[wave][r] = rs[v];
    }
  }
  __syncthreads();

  if (threadIdx.x < 16) {
    float M = -3.0e38f;
    for (int w = 0; w < 8; ++w) M = fmaxf(M, wmaxs[w][threadIdx.x]);
    float S = 0.f;
    for (int w = 0; w < 8; ++w) S += wsums[w][threadIdx.x] * __expf(wmaxs[w][threadIdx.x] - M);
    gmax[threadIdx.x] = M;
    gsum[threadIdx.x] = S;
  }
  __syncthreads();

  // combine per-wave partial outputs into O_sh (LDS float atomics)
  float fcomb[8];
#pragma unroll
  for (int v = 0; v < 8; ++v) fcomb[v] = __expf(rm[v] - gmax[v + 8 * half]);
#pragma unroll
  for (int t = 0; t < 8; ++t)
#pragma unroll
    for (int v = 0; v < 8; ++v) {
      const int r = v + 8 * half;
      atomicAdd(&O_sh[r * DDIM + t * 16 + l16], o[t][v] * fcomb[v]);
    }
  __syncthreads();

  // squared distance vs Qv, masked to valid p rows, scaled by 1/(h*w)
  float acc = 0.f;
  for (int e = threadIdx.x; e < 16 * DDIM; e += 256) {
    const int r = e >> 7, d = e & 127;
    const int p = pt * 16 + r;
    if (p < PVAL) {
      float ov = O_sh[e] / gsum[r];
      float qv = Qv[((size_t)b * PPAD + p) * DDIM + d];
      float df = qv - ov;
      acc += df * df;
    }
  }
  acc *= (1.0f / (float)PVAL);
#pragma unroll
  for (int d = 16; d >= 1; d >>= 1) acc += __shfl_xor(acc, d, 32);
  if (lane == 0) wred[wave] = acc;
  __syncthreads();
  if (threadIdx.x == 0) {
    float t = 0.f;
    for (int w = 0; w < 8; ++w) t += wred[w];
    atomicAdd(out_scalar, t);
  }
}

// ---------------- host-side launch ----------------

extern "C" void kernel_launch(void* const* d_in, const int* in_sizes, int n_in,
                              void* d_out, int out_size, void* d_ws, size_t ws_size,
                              hipStream_t stream) {
  const float* query    = (const float*)d_in[0];
  const float* supports = (const float*)d_in[1];
  const float* W_qk     = (const float*)d_in[2];
  const float* W_v      = (const float*)d_in[3];

  char* ws = (char*)d_ws;
  size_t off = 0;
  auto walloc = [&](size_t bytes) -> void* {
    void* p = ws + off;
    off += (bytes + 255) & ~(size_t)255;
    return p;
  };
  const size_t qb_bytes = (size_t)BATCH * PPAD * CDIM * 2;
  const size_t sb_bytes = (size_t)BATCH * MPAD * CDIM * 2;
  u16*   Qb  = (u16*)walloc(qb_bytes);
  u16*   Sb  = (u16*)walloc(sb_bytes);
  u16*   WkB = (u16*)walloc((size_t)DDIM * CDIM * 2);
  u16*   WvB = (u16*)walloc((size_t)DDIM * CDIM * 2);
  u16*   Qq  = (u16*)walloc((size_t)BATCH * PPAD * DDIM * 2);
  float* Qv  = (float*)walloc((size_t)BATCH * PPAD * DDIM * 4);
  u16*   Sk  = (u16*)walloc((size_t)BATCH * MPAD * DDIM * 2);
  u16*   Sv  = (u16*)walloc((size_t)BATCH * DDIM * MPAD * 2);

  hipMemsetAsync(d_out, 0, sizeof(float), stream);
  hipMemsetAsync(Qb, 0, qb_bytes, stream);

  zero_tail_kernel<<<128, 256, 0, stream>>>(Sb);
  conv_s_kernel<<<dim3(16, NSUP, BATCH), 256, 0, stream>>>(supports, Sb);
  {
    size_t tot = (size_t)BATCH * CDIM * 49;
    conv_q_kernel<<<(unsigned)((tot + 255) / 256), 256, 0, stream>>>(query, Qb);
  }
  conv_w_kernel<<<128, 256, 0, stream>>>(W_qk, W_v, WkB, WvB);

  proj_supports_kernel<<<dim3(41, 8, BATCH), 256, 0, stream>>>(Sb, WkB, WvB, Sk, Sv);
  proj_query_kernel<<<dim3(13, 8, BATCH), 64, 0, stream>>>(Qb, WkB, WvB, Qq, Qv);
  attn_out_kernel<<<dim3(13, BATCH), 256, 0, stream>>>(Qq, Qv, Sk, Sv, (float*)d_out);
}
